// g_s_38972533243997
// MI455X (gfx1250) — 8-bit matrix operand order repaired and checked by reading, compiled, not run on hardware
//
#include <hip/hip_runtime.h>


typedef __attribute__((ext_vector_type(8))) int v8i;
typedef unsigned char u8;
typedef signed char   i8;

__device__ __forceinline__ int tapN(int b){ return b ? 2 : 3; }
__device__ __forceinline__ int tapK(int b, int i){ return b ? 3-2*i : 4-2*i; }
__device__ __forceinline__ int tapD(int b, int i){ return b ? i : i-1; }

__device__ __forceinline__ void async_b32_to_lds(int* ldsDst, const int* gSrc){
  unsigned ldsOff = (unsigned)(unsigned long long)ldsDst;
  asm volatile("global_load_async_to_lds_b32 %0, %1, off"
               :: "v"(ldsOff), "v"(gSrc) : "memory");
}
__device__ __forceinline__ void wait_async2(){  asm volatile("s_wait_asynccnt 2"  ::: "memory"); }
__device__ __forceinline__ void wait_async16(){ asm volatile("s_wait_asynccnt 16" ::: "memory"); }


__global__ void zero_kernel(int* __restrict__ p, int n){
  int i = blockIdx.x*256 + threadIdx.x;
  if (i < n) p[i] = 0;
}

__global__ void quant_x_kernel(const float* __restrict__ x, u8* __restrict__ q, int n){
  int i = blockIdx.x*256 + threadIdx.x;
  if (i >= n) return;
  const int c = i / 2304, rem = i - c*2304;
  const int y = rem / 48, xx = rem - y*48;
  int v = (int)rintf(x[i]); v = v<0?0:(v>255?255:v);
  q[(((((c>>2)*50) + (y+1))*50 + (xx+1)) << 2) + (c & 3)] = (u8)v;
}

__global__ void pack_b_kernel(const float* __restrict__ b, int* __restrict__ qb, int n, int npad){
  int i = blockIdx.x*64 + threadIdx.x;
  if (i < npad) qb[i] = (i < n) ? (int)rintf(b[i]) : 0;
}

__global__ void pack_w_kernel(const float* __restrict__ w, i8* __restrict__ dst,
                              int IC, int OCsrc, int nK, int py, int px, int total)
{
  int idx = blockIdx.x*256 + threadIdx.x;
  if (idx >= total) return;
  const int bb   = idx & 3;
  const int r    = (idx >> 2) & 7;
  const int lane = (idx >> 5) & 31;
  const int rest = idx >> 10;
  const int k64  = rest % nK;
  const int oc_t = rest / nK;
  const int oc   = (oc_t << 4) + (lane & 15);
  const int Koff = ((r >= 4) ? 32 : 0) + (((r & 3) >= 2) ? 8 : 0) +
                   ((lane >= 16) ? 16 : 0) + ((r & 1) << 2) + bb;
  const int Kk = (k64 << 6) + Koff;
  const int ntx = tapN(px);
  const int t  = Kk / IC, ic = Kk - t*IC;
  const int ty = t / ntx, tx = t - ty*ntx;
  const int ky = tapK(py, ty), kx = tapK(px, tx);
  i8 v = 0;
  if (oc < OCsrc) {
    const float wv = w[(((long long)ic*OCsrc + oc)*5 + ky)*5 + kx];
    int q = (int)rintf(wv);
    q = q < -128 ? -128 : (q > 127 ? 127 : q);
    v = (i8)q;
  }
  dst[idx] = v;
}

__device__ __forceinline__ int stage_advance(int& s_ic, int& s_tx,
                                             int ICc, int ntx, int Wp, int HpW16)
{
  int dlt;
  if (++s_ic != ICc) dlt = HpW16;
  else {
    s_ic = 0;
    if (++s_tx != ntx) dlt = 1 - (ICc - 1)*HpW16;
    else { s_tx = 0; dlt = Wp - (ntx - 1) - (ICc - 1)*HpW16; }
  }
  return dlt;
}


__global__ void __launch_bounds__(384)
deconv_mid_kernel(const int* __restrict__ in4, const i8* __restrict__ Apack,
                  const int* __restrict__ qb, int* __restrict__ out4,
                  int IC, int Hin, int Win, int layer,
                  const float* __restrict__ muls, const float* __restrict__ relus,
                  const int* __restrict__ dvds, const int* __restrict__ bitsPtr)
{
  __shared__ int bt[4][256];
  const int tid = threadIdx.x, lane = tid & 31, wvid = tid >> 5;
  const int p = blockIdx.z, py = p >> 1, px = p & 1;
  const int nty = tapN(py), ntx = tapN(px);
  const int nK = (nty*ntx*IC) >> 6;
  long long parOff = 0;
  {
    const int tcnt[4] = {9,6,6,4};
    for (int q = 0; q < p; ++q) parOff += 192LL * tcnt[q] * IC;
  }
  const i8* Ab = Apack + parOff + (long long)wvid * nK * 1024 + lane*32;
  const int gy = blockIdx.y, gx0 = blockIdx.x << 4;
  const int Hp = Hin + 2, Wp = Win + 2;
  const int HpW16 = Hp*Wp*16, ICc = IC >> 6;
  const bool producer = (__builtin_amdgcn_readfirstlane(wvid) < 8);

  const int lane_t = tid & 31, r_t = tid >> 5;
  const int cgOff = ((((r_t >= 4) ? 32 : 0) + ((lane_t >= 16) ? 16 : 0) + ((r_t & 3) << 2)) >> 2);
  const int nst = lane_t & 15;

  const int* pst = in4 + (cgOff*Hp + (gy + tapD(py, 0) + 1))*Wp
                       + (gx0 + tapD(px, 0) + nst + 1);
  int s_ic = 0, s_tx = 0, st = 0;

  #pragma unroll
  for (int i0 = 0; i0 < 3; ++i0) {
    if (producer) async_b32_to_lds(&bt[i0][tid], pst);
    pst += stage_advance(s_ic, s_tx, ICc, ntx, Wp, HpW16);
    ++st;
  }
  wait_async2();
  __syncthreads();

  v8i acc = {0,0,0,0,0,0,0,0};

  #pragma unroll 2
  for (int k = 0; k < nK; ++k) {
    if (producer) async_b32_to_lds(&bt[(k + 3) & 3][tid], pst);
    if (st < nK - 1) { pst += stage_advance(s_ic, s_tx, ICc, ntx, Wp, HpW16); ++st; }
    const v8i a = *reinterpret_cast<const v8i*>(Ab + (long long)k * 1024);
    v8i b;
    #pragma unroll
    for (int i = 0; i < 8; ++i) b[i] = bt[k & 3][i*32 + lane];
    acc = __builtin_amdgcn_wmma_i32_16x16x64_iu8(true, a, false, b, acc, false, false);
    wait_async2();
    __syncthreads();
  }

  const int   dv   = dvds[layer];
  const float relu = relus[layer];
  const int sks[3] = {3,4,3};
  const int sk = sks[layer];
  const float maxq = exp2f((float)bitsPtr[0]) - 1.0f;
  const long long clp = (long long)rintf(maxq / relu * 33554432.0f);
  const long long scl = (long long)floorf((relu + exp2f((float)(sk-1))) * exp2f((float)(-sk)));
  const int Hpo = (Hin << 1) + 2, Wpo = (Win << 1) + 2;
  const int nl = lane & 15, hi8 = (lane >> 4) << 3;
  const int oy = (gy << 1) + py;
  const int ox = ((gx0 + nl) << 1) + px;
  const int sh1 = dv - 9;
  unsigned dw[2] = {0u, 0u};
  #pragma unroll
  for (int r = 0; r < 8; ++r) {
    const int oc = (wvid << 4) + hi8 + r;
    const long long m  = (long long)(int)rintf(muls[oc]);
    const long long t0 = (long long)(acc[r] + qb[oc]) * m;
    long long r1 = (t0 + (1LL << (sh1-1))) >> sh1;
    r1 = r1 < 0 ? 0 : (r1 > clp ? clp : r1);
    long long r2 = (r1 * scl + (1LL << (24 - sk))) >> (25 - sk);
    r2 = r2 < 0 ? 0 : (r2 > 255 ? 255 : r2);
    dw[r >> 2] |= (unsigned)r2 << ((r & 3) * 8);
  }
  const int cgb = ((wvid << 4) + hi8) >> 2;
  int* o4 = out4 + (cgb*Hpo + (oy+1))*Wpo + (ox+1);
  o4[0]        = (int)dw[0];
  o4[Hpo*Wpo]  = (int)dw[1];
}


__global__ void __launch_bounds__(32)
deconv_last_kernel(const int* __restrict__ in4, const i8* __restrict__ Apack,
                   const int* __restrict__ qb, float* __restrict__ out,
                   int IC, int Hin, int Win,
                   const float* __restrict__ muls, const int* __restrict__ dvds)
{
  __shared__ int bt[4][256];
  const int lane = threadIdx.x & 31;
  const int p = blockIdx.z, py = p >> 1, px = p & 1;
  const int nty = tapN(py), ntx = tapN(px);
  const int nK = (nty*ntx*IC) >> 6;
  long long parOff = 0;
  {
    const int tcnt[4] = {9,6,6,4};
    for (int q = 0; q < p; ++q) parOff += 16LL * tcnt[q] * IC;
  }
  const i8* Ab = Apack + parOff + lane*32;
  const int gy = blockIdx.y, gx0 = blockIdx.x << 4;
  const int Hp = Hin + 2, Wp = Win + 2;
  const int HpW = Hp*Wp, HpW16 = HpW*16, ICc = IC >> 6;
  const int nst = lane & 15;

  const int* pst = in4 + ((((lane >= 16) ? 4 : 0))*Hp + (gy + tapD(py, 0) + 1))*Wp
                       + (gx0 + tapD(px, 0) + nst + 1);
  int s_ic = 0, s_tx = 0, st = 0;

  #pragma unroll
  for (int i0 = 0; i0 < 3; ++i0) {
    #pragma unroll
    for (int j = 0; j < 8; ++j) {
      const int cadd = ((j & 3) + ((j >= 4) ? 8 : 0)) * HpW;
      async_b32_to_lds(&bt[i0][j*32 + lane], pst + cadd);
    }
    pst += stage_advance(s_ic, s_tx, ICc, ntx, Wp, HpW16);
    ++st;
  }
  wait_async16();
  __syncthreads();

  v8i acc = {0,0,0,0,0,0,0,0};

  #pragma unroll 2
  for (int k = 0; k < nK; ++k) {
    #pragma unroll
    for (int j = 0; j < 8; ++j) {
      const int cadd = ((j & 3) + ((j >= 4) ? 8 : 0)) * HpW;
      async_b32_to_lds(&bt[(k + 3) & 3][j*32 + lane], pst + cadd);
    }
    if (st < nK - 1) { pst += stage_advance(s_ic, s_tx, ICc, ntx, Wp, HpW16); ++st; }
    const v8i a = *reinterpret_cast<const v8i*>(Ab + (long long)k * 1024);
    v8i b;
    #pragma unroll
    for (int i = 0; i < 8; ++i) b[i] = bt[k & 3][i*32 + lane];
    acc = __builtin_amdgcn_wmma_i32_16x16x64_iu8(true, a, false, b, acc, false, false);
    wait_async16();
    __syncthreads();
  }

  const int dv = dvds[3];
  const int Hout = Hin << 1, Wout = Win << 1;
  const int nl = lane & 15, hi8 = (lane >> 4) << 3;
  const int oy = (gy << 1) + py;
  const long long obase = (long long)oy * Wout + (long long)((gx0 + nl) << 1) + px;
  #pragma unroll
  for (int r = 0; r < 8; ++r) {
    const int oc = hi8 + r;
    if (oc < 3) {
      const long long m  = (long long)(int)rintf(muls[oc]);
      const long long t0 = (long long)(acc[r] + qb[oc]) * m;
      const long long r1 = (t0 + (1LL << (dv-9))) >> (dv-8);
      out[(long long)oc * Hout * Wout + obase] = (float)r1 / 255.0f;
    }
  }
}


extern "C" void kernel_launch(void* const* d_in, const int* in_sizes, int n_in,
                              void* d_out, int out_size, void* d_ws, size_t ws_size,
                              hipStream_t stream)
{
  (void)in_sizes; (void)n_in; (void)out_size; (void)ws_size;
  const float* x = (const float*)d_in[0];
  const float* W[4] = {(const float*)d_in[1], (const float*)d_in[3],
                       (const float*)d_in[5], (const float*)d_in[7]};
  const float* B[4] = {(const float*)d_in[2], (const float*)d_in[4],
                       (const float*)d_in[6], (const float*)d_in[8]};
  const float* M[4] = {(const float*)d_in[9], (const float*)d_in[10],
                       (const float*)d_in[11], (const float*)d_in[12]};
  const float* relus = (const float*)d_in[13];
  const int*   dvds  = (const int*)d_in[14];
  const int*   bits  = (const int*)d_in[15];

  const int ICs[4] = {320,192,192,192};
  const int OCs[4] = {192,192,192,3};
  const int OCp[4] = {192,192,192,16};

  char* ws = (char*)d_ws;
  size_t off = 0;
  auto alloc = [&](size_t sz){ size_t o = off; off += (sz + 255) & ~(size_t)255; return o; };
  const size_t qx_o = alloc((size_t)80*50*50*4);
  const size_t a1_o = alloc((size_t)48*98*98*4);
  const size_t a2_o = alloc((size_t)48*194*194*4);
  const size_t a3_o = alloc((size_t)48*386*386*4);
  size_t ap_o[4], qb_o[4];
  for (int l = 0; l < 4; ++l) ap_o[l] = alloc((size_t)OCp[l]*25*ICs[l]);
  for (int l = 0; l < 4; ++l) qb_o[l] = alloc((size_t)OCp[l]*4);

  {
    const size_t zo[4] = {qx_o, a1_o, a2_o, a3_o};
    const int    zn[4] = {80*50*50, 48*98*98, 48*194*194, 48*386*386};
    for (int l = 0; l < 4; ++l)
      zero_kernel<<<(zn[l]+255)/256, 256, 0, stream>>>((int*)(ws+zo[l]), zn[l]);
  }

  { int n = 320*48*48;
    quant_x_kernel<<<(n+255)/256, 256, 0, stream>>>(x, (u8*)(ws+qx_o), n); }

  for (int l = 0; l < 4; ++l) {
    size_t po = 0;
    for (int p = 0; p < 4; ++p) {
      const int py = p >> 1, px = p & 1;
      const int taps = (py ? 2 : 3) * (px ? 2 : 3);
      const int Kp = taps * ICs[l];
      const int nK = Kp >> 6;
      const int total = OCp[l] * Kp;
      pack_w_kernel<<<(total+255)/256, 256, 0, stream>>>(
          W[l], (i8*)(ws+ap_o[l]+po), ICs[l], OCs[l], nK, py, px, total);
      po += (size_t)OCp[l] * Kp;
    }
    pack_b_kernel<<<(OCp[l]+63)/64, 64, 0, stream>>>(B[l], (int*)(ws+qb_o[l]), OCs[l], OCp[l]);
  }

  const int* in0 = (const int*)(ws+qx_o);
  int* a1 = (int*)(ws+a1_o); int* a2 = (int*)(ws+a2_o); int* a3 = (int*)(ws+a3_o);

  deconv_mid_kernel<<<dim3(3,48,4),   384, 0, stream>>>(in0, (const i8*)(ws+ap_o[0]),
      (const int*)(ws+qb_o[0]), a1, 320,  48,  48, 0, M[0], relus, dvds, bits);
  deconv_mid_kernel<<<dim3(6,96,4),   384, 0, stream>>>(a1, (const i8*)(ws+ap_o[1]),
      (const int*)(ws+qb_o[1]), a2, 192,  96,  96, 1, M[1], relus, dvds, bits);
  deconv_mid_kernel<<<dim3(12,192,4), 384, 0, stream>>>(a2, (const i8*)(ws+ap_o[2]),
      (const int*)(ws+qb_o[2]), a3, 192, 192, 192, 2, M[2], relus, dvds, bits);
  deconv_last_kernel<<<dim3(24,384,4), 32, 0, stream>>>(a3, (const i8*)(ws+ap_o[3]),
      (const int*)(ws+qb_o[3]), (float*)d_out, 192, 384, 384, M[3], dvds);
}
